// MultiHeadAttention_14929306321720
// MI455X (gfx1250) — compile-verified, matrix operand order checked by reading, not run on hardware
//
#include <hip/hip_runtime.h>
#include <hip/hip_bf16.h>

// MI455X (gfx1250) fused multi-head attention, bf16 WMMA + f32 accumulate.
// B=2, S=2048, D=1024, H=16, depth=64.

typedef __bf16 bf16;
typedef __attribute__((ext_vector_type(16))) __bf16 v16bf;
typedef __attribute__((ext_vector_type(8)))  __bf16 v8bf;
typedef __attribute__((ext_vector_type(8)))  float  v8f;
typedef unsigned int uint32x4 __attribute__((ext_vector_type(4)));
typedef int          int32x4  __attribute__((ext_vector_type(4)));
typedef int          int32x8  __attribute__((ext_vector_type(8)));

#define BB 2
#define SS 2048
#define DD 1024
#define HH 16
#define DEPTH 64

#if defined(__has_builtin)
#if __has_builtin(__builtin_amdgcn_tensor_load_to_lds)
#define HAVE_TDM 1
#endif
#endif

union FragU { v16bf v; v8bf h[2]; __bf16 e[16]; };

__device__ __forceinline__ v8f wmma_bf16(v16bf a, v16bf b, v8f c) {
  return __builtin_amdgcn_wmma_f32_16x16x32_bf16(false, a, false, b, (short)0, c,
                                                 false, false);
}

// Fragment loader where the contraction axis is contiguous bf16:
// each lane needs two contiguous 8-element runs (16 B each) at kb0/kb1.
__device__ __forceinline__ v16bf frag_ld_bf16(const bf16* p, int kb0, int kb1) {
  FragU f;
  f.h[0] = *(const v8bf*)(p + kb0);
  f.h[1] = *(const v8bf*)(p + kb1);
  return f.v;
}

// fp32 source (LDS probability strip) -> bf16 fragment.
__device__ __forceinline__ v16bf frag_cvt_f32(const float* p, int kb0, int kb1) {
  FragU f;
#pragma unroll
  for (int j = 0; j < 8; ++j) f.e[j] = (__bf16)p[kb0 + j];
#pragma unroll
  for (int j = 0; j < 8; ++j) f.e[8 + j] = (__bf16)p[kb1 + j];
  return f.v;
}

// Flat LDS pointer -> 32-bit LDS offset (flat LDS addr low 32 bits are the
// in-allocation offset per CDNA5 aperture mapping).
__device__ __forceinline__ unsigned int lds_off32(const void* p) {
  return (unsigned int)(unsigned long long)p;
}

__device__ __forceinline__ void async_ld_b128(unsigned int lds_off,
                                              const void* gaddr) {
  asm volatile("global_load_async_to_lds_b128 %0, %1, off"
               :: "v"(lds_off), "v"((unsigned long long)gaddr) : "memory");
}
__device__ __forceinline__ void wait_async0() {
  asm volatile("s_wait_asynccnt 0x0" ::: "memory");
}

// ---------------------------------------------------------------------------
// fp32 -> bf16 bulk convert (8 elements / thread, 16 B stores).
// ---------------------------------------------------------------------------
__global__ __launch_bounds__(256) void cvt_kernel(const float* __restrict__ src,
                                                  bf16* __restrict__ dst, int n8) {
  int i = blockIdx.x * 256 + threadIdx.x;
  if (i >= n8) return;
  const float* s = src + (size_t)i * 8;
  union { v8bf v; __bf16 e[8]; } o;
#pragma unroll
  for (int j = 0; j < 8; ++j) o.e[j] = (__bf16)s[j];
  *(v8bf*)(dst + (size_t)i * 8) = o.v;
}

// ---------------------------------------------------------------------------
// Projection GEMM: Out = X(bf16)[4096,1024] * W(bf16)[1024,1024]^T + bias,
// output bf16 head-split. vmode==0 -> [B,H,S,64]; vmode==1 -> [B,H,64,S].
// 256 threads (8 waves), tile 128(M) x 64(N). Weight tiles double-buffered in
// LDS via GLOBAL_LOAD_ASYNC_TO_LDS_B128: chunk i+1 DMA overlaps chunk i WMMAs.
// ---------------------------------------------------------------------------
__global__ __launch_bounds__(256) void proj_kernel(const bf16* __restrict__ X,
                                                   const bf16* __restrict__ W,
                                                   const float* __restrict__ bias,
                                                   bf16* __restrict__ Out,
                                                   int vmode) {
  __shared__ bf16 Wt[2][64 * 32];
  const int tid  = threadIdx.x;
  const int lane = tid & 31, wave = tid >> 5;
  const int half = lane >> 4, r16 = lane & 15;
  const int kb0 = half * 8, kb1 = 16 + half * 8;
  const int n0 = blockIdx.x * 64;
  const int m_base = blockIdx.y * 128 + wave * 16;

  const int nl = tid >> 2;             // 0..63 : W row inside tile
  const int kl = (tid & 3) * 8;        // 16 B chunk inside row
  const unsigned int wt_off[2] = { lds_off32(&Wt[0][nl * 32 + kl]),
                                   lds_off32(&Wt[1][nl * 32 + kl]) };
  const bf16* wrow = W + (size_t)(n0 + nl) * DD + kl;

  v8f acc[4] = {};

  async_ld_b128(wt_off[0], wrow);      // pre-issue chunk 0
  for (int i = 0; i < DD / 32; ++i) {
    const int k0 = i * 32;
    const int cur = i & 1;
    wait_async0();                     // our b128s for buf[cur] landed
    __syncthreads();                   // everyone's landed; prev reads done
    if (i + 1 < DD / 32)               // overlap next DMA with this compute
      async_ld_b128(wt_off[cur ^ 1], wrow + k0 + 32);

    v16bf a = frag_ld_bf16(X + (size_t)(m_base + r16) * DD + k0, kb0, kb1);
    v16bf bfr[4];
#pragma unroll
    for (int nt = 0; nt < 4; ++nt)
      bfr[nt] = frag_ld_bf16(&Wt[cur][(nt * 16 + r16) * 32], kb0, kb1);
#pragma unroll
    for (int nt = 0; nt < 4; ++nt)
      acc[nt] = wmma_bf16(a, bfr[nt], acc[nt]);
  }

#pragma unroll
  for (int nt = 0; nt < 4; ++nt) {
    int n = n0 + nt * 16 + r16;
    int h = n >> 6, dh = n & 63;
    float bv = bias[n];
#pragma unroll
    for (int r = 0; r < 8; ++r) {
      int m = m_base + half * 8 + r;     // C/D layout: M = r + 8*half
      int b = m >> 11, s = m & (SS - 1);
      float val = acc[nt][r] + bv;
      size_t idx = vmode ? ((((size_t)b * HH + h) * DEPTH + dh) * SS + s)
                         : ((((size_t)b * HH + h) * SS + s) * DEPTH + dh);
      Out[idx] = (__bf16)val;
    }
  }
}

// ---------------------------------------------------------------------------
// Fused attention: per (b,h), per 32-query strip:
//   logits (WMMA over d) -> 32x2048 fp32 LDS strip -> softmax(+mask) ->
//   single-pass attn write -> P@V (WMMA) -> ctx.
// ---------------------------------------------------------------------------
__global__ __launch_bounds__(256) void attn_kernel(const bf16* __restrict__ Qh,
                                                   const bf16* __restrict__ Kh,
                                                   const bf16* __restrict__ Vt,
                                                   const float* __restrict__ mask,
                                                   float* __restrict__ attn,
                                                   bf16* __restrict__ Ctx) {
  extern __shared__ float smem[];
  float* P       = smem;                 // [32][2048]
  float* red     = smem + 32 * 2048;     // [32][8]
  float* rowstat = red + 256;            // [32]

  const int tid  = threadIdx.x;
  const int lane = tid & 31, wave = tid >> 5;
  const int half = lane >> 4, r16 = lane & 15;
  const int kb0 = half * 8, kb1 = 16 + half * 8;
  const int bh = blockIdx.y;
  const int b_idx = bh >> 4;
  const int qbase = blockIdx.x * 32;
  const size_t head_off = (size_t)bh * SS * DEPTH;

  // ---- Phase 1: logits strip. Each wave owns 256 key columns. ----
  v16bf aq[2][2];
#pragma unroll
  for (int mt = 0; mt < 2; ++mt)
#pragma unroll
    for (int kc = 0; kc < 2; ++kc)
      aq[mt][kc] = frag_ld_bf16(
          Qh + head_off + (size_t)(qbase + mt * 16 + r16) * DEPTH + kc * 32,
          kb0, kb1);

  const float scale = 0.125f;  // 1/sqrt(64)
  const int keyw = wave * 256;
  for (int nt = 0; nt < 16; ++nt) {
    int key = keyw + nt * 16 + r16;
    const bf16* kp = Kh + head_off + (size_t)key * DEPTH;
    v16bf bk0 = frag_ld_bf16(kp, kb0, kb1);
    v16bf bk1 = frag_ld_bf16(kp + 32, kb0, kb1);
    float madd = mask[b_idx * SS + key] * -1e9f;
#pragma unroll
    for (int mt = 0; mt < 2; ++mt) {
      v8f acc = {};
      acc = wmma_bf16(aq[mt][0], bk0, acc);
      acc = wmma_bf16(aq[mt][1], bk1, acc);
#pragma unroll
      for (int r = 0; r < 8; ++r)
        P[(mt * 16 + half * 8 + r) * 2048 + key] = acc[r] * scale + madd;
    }
  }
  __syncthreads();

  // ---- Phase 2: row softmax (8 threads/row) + prefetch V^T slab for P@V. ----
  {
    const bf16* vslab = Vt + (size_t)bh * DEPTH * SS;
    for (int off = tid * 128; off < DEPTH * SS; off += 256 * 128)
      __builtin_prefetch(vslab + off, 0, 3);  // global_prefetch_b8 -> warm L2

    int row = tid >> 3, seg = tid & 7;
    float* pr = P + row * 2048;
    int c0 = seg * 256;
    float mx = -1e30f;
    for (int c = 0; c < 256; ++c) mx = fmaxf(mx, pr[c0 + c]);
    red[row * 8 + seg] = mx;
    __syncthreads();
    if (seg == 0) {
      float m2 = red[row * 8];
      for (int j = 1; j < 8; ++j) m2 = fmaxf(m2, red[row * 8 + j]);
      rowstat[row] = m2;
    }
    __syncthreads();
    float rm = rowstat[row];
    float sum = 0.f;
    for (int c = 0; c < 256; ++c) {
      float e = __expf(pr[c0 + c] - rm);
      pr[c0 + c] = e;
      sum += e;
    }
    red[row * 8 + seg] = sum;
    __syncthreads();
    if (seg == 0) {
      float s2 = 0.f;
      for (int j = 0; j < 8; ++j) s2 += red[row * 8 + j];
      rowstat[row] = 1.0f / s2;
    }
    __syncthreads();
  }

  // ---- Normalize + single coalesced write of attn. ----
  {
    float* ao = attn + (size_t)bh * SS * SS + (size_t)qbase * SS;
    for (int idx = tid; idx < 32 * 2048; idx += 256) {
      int r = idx >> 11;
      float v = P[idx] * rowstat[r];
      P[idx] = v;
      ao[(size_t)r * SS + (idx & 2047)] = v;
    }
  }
  __syncthreads();

  // ---- Phase 3: ctx strip = P(32x2048) @ V(2048x64). One 16x16 tile/wave. ----
  {
    const int mt = wave >> 2, nt = wave & 3;
    const int dh = nt * 16 + r16;
    const bf16* vp = Vt + (size_t)bh * DEPTH * SS + (size_t)dh * SS;  // V^T row
    v8f acc = {};
    for (int kc = 0; kc < SS; kc += 32) {
      v16bf a = frag_cvt_f32(P + (mt * 16 + r16) * 2048 + kc, kb0, kb1);
      v16bf b = frag_ld_bf16(vp + kc, kb0, kb1);
      acc = wmma_bf16(a, b, acc);
    }
    const int head = bh & 15;
#pragma unroll
    for (int r = 0; r < 8; ++r) {
      int m = qbase + mt * 16 + half * 8 + r;
      Ctx[((size_t)b_idx * SS + m) * DD + head * DEPTH + dh] = (__bf16)acc[r];
    }
  }
}

// ---------------------------------------------------------------------------
// Output projection: out = Ctx(bf16) * dense_w(bf16)^T + dense_b (fp32 out).
// Weight tiles double-buffered via the Tensor Data Mover (TENSORcnt) when
// available; the next chunk's D# is issued before the current compute phase.
// ---------------------------------------------------------------------------
#if defined(HAVE_TDM)
__device__ __forceinline__ void tdm_issue_wtile(const bf16* tile_start,
                                                unsigned int lds_addr) {
  unsigned long long ga = (unsigned long long)tile_start;
  uint32x4 g0;
  g0[0] = 1u;                                      // count=1, user mode
  g0[1] = lds_addr;                                // lds_addr
  g0[2] = (unsigned int)ga;                        // global_addr[31:0]
  g0[3] = (unsigned int)(ga >> 32) | 0x80000000u;  // addr[56:32] | type=2
  int32x8 g1;
  g1[0] = 0x10000;                                 // data_size=1 (2 bytes)
  g1[1] = (int)((DD & 0xFFFF) << 16);              // tensor_dim0 = 1024
  g1[2] = (int)((DD >> 16) | ((DD & 0xFFFF) << 16));  // tensor_dim1 = 1024
  g1[3] = (int)((DD >> 16) | (32u << 16));         // tile_dim0 = 32 (k)
  g1[4] = 64;                                      // tile_dim1 = 64 (n)
  g1[5] = DD;                                      // tensor_dim0_stride = 1024
  g1[6] = 0;
  g1[7] = 0;
  int32x4 gz = {0, 0, 0, 0};
#if __clang_major__ >= 23
  int32x8 gz8 = {0, 0, 0, 0, 0, 0, 0, 0};
  __builtin_amdgcn_tensor_load_to_lds(g0, g1, gz, gz, gz8, 0);
#else
  __builtin_amdgcn_tensor_load_to_lds(g0, g1, gz, gz, 0);
#endif
}
#endif

__global__ __launch_bounds__(256) void dense_kernel(const bf16* __restrict__ Xc,
                                                    const bf16* __restrict__ W,
                                                    const float* __restrict__ bias,
                                                    float* __restrict__ Out) {
  __shared__ bf16 Wt[2][64 * 32];
  const int tid  = threadIdx.x;
  const int lane = tid & 31, wave = tid >> 5;
  const int half = lane >> 4, r16 = lane & 15;
  const int kb0 = half * 8, kb1 = 16 + half * 8;
  const int n0 = blockIdx.x * 64;
  const int m_base = blockIdx.y * 128 + wave * 16;

  v8f acc[4] = {};

#if defined(HAVE_TDM)
  const bf16* wbase = W + (size_t)n0 * DD;
  const unsigned int wt_lds[2] = { lds_off32(&Wt[0][0]), lds_off32(&Wt[1][0]) };
  if (tid < 32) tdm_issue_wtile(wbase, wt_lds[0]);   // pre-issue chunk 0
  for (int i = 0; i < DD / 32; ++i) {
    const int k0 = i * 32;
    const int cur = i & 1;
    if (tid < 32) __builtin_amdgcn_s_wait_tensorcnt(0);
    __syncthreads();
    if (tid < 32 && i + 1 < DD / 32)
      tdm_issue_wtile(wbase + k0 + 32, wt_lds[cur ^ 1]);
#else
  const int nl = tid >> 2;
  const int kl = (tid & 3) * 8;
  const unsigned int wt_off[2] = { lds_off32(&Wt[0][nl * 32 + kl]),
                                   lds_off32(&Wt[1][nl * 32 + kl]) };
  const bf16* wrow = W + (size_t)(n0 + nl) * DD + kl;
  async_ld_b128(wt_off[0], wrow);
  for (int i = 0; i < DD / 32; ++i) {
    const int k0 = i * 32;
    const int cur = i & 1;
    wait_async0();
    __syncthreads();
    if (i + 1 < DD / 32) async_ld_b128(wt_off[cur ^ 1], wrow + k0 + 32);
#endif
    v16bf a = frag_ld_bf16(Xc + (size_t)(m_base + r16) * DD + k0, kb0, kb1);
    v16bf bfr[4];
#pragma unroll
    for (int nt = 0; nt < 4; ++nt)
      bfr[nt] = frag_ld_bf16(&Wt[cur][(nt * 16 + r16) * 32], kb0, kb1);
#pragma unroll
    for (int nt = 0; nt < 4; ++nt)
      acc[nt] = wmma_bf16(a, bfr[nt], acc[nt]);
  }

#pragma unroll
  for (int nt = 0; nt < 4; ++nt) {
    int n = n0 + nt * 16 + r16;
    float bv = bias[n];
#pragma unroll
    for (int r = 0; r < 8; ++r) {
      int m = m_base + half * 8 + r;
      Out[(size_t)m * DD + n] = acc[nt][r] + bv;
    }
  }
}

extern "C" void kernel_launch(void* const* d_in, const int* in_sizes, int n_in,
                              void* d_out, int out_size, void* d_ws, size_t ws_size,
                              hipStream_t stream) {
  const float* q       = (const float*)d_in[0];
  const float* k       = (const float*)d_in[1];
  const float* v       = (const float*)d_in[2];
  const float* mask    = (const float*)d_in[3];
  const float* wq_w    = (const float*)d_in[4];
  const float* wq_b    = (const float*)d_in[5];
  const float* wk_w    = (const float*)d_in[6];
  const float* wk_b    = (const float*)d_in[7];
  const float* wv_w    = (const float*)d_in[8];
  const float* wv_b    = (const float*)d_in[9];
  const float* dense_w = (const float*)d_in[10];
  const float* dense_b = (const float*)d_in[11];

  const size_t HEADSZ = (size_t)BB * HH * SS * DEPTH;  // 4,194,304
  const size_t WSZ    = (size_t)DD * DD;               // 1,048,576
  bf16* Qh  = (bf16*)d_ws;          // projected Q  [B,H,S,64]
  bf16* Kh  = Qh + HEADSZ;          // projected K  [B,H,S,64]
  bf16* Vt  = Kh + HEADSZ;          // projected V^T [B,H,64,S]
  bf16* Ctx = Vt + HEADSZ;          // context [B,S,D]
  bf16* Qb  = Ctx + HEADSZ;         // bf16 copies of activations
  bf16* Kb  = Qb + HEADSZ;
  bf16* Vb  = Kb + HEADSZ;
  bf16* Wqb = Vb + HEADSZ;          // bf16 copies of weights
  bf16* Wkb = Wqb + WSZ;
  bf16* Wvb = Wkb + WSZ;
  bf16* Wdb = Wvb + WSZ;

  float* out  = (float*)d_out;                       // [2,2048,1024]
  float* attn = out + (size_t)BB * SS * DD;          // [2,16,2048,2048]

  dim3 blk(256);
  cvt_kernel<<<2048, blk, 0, stream>>>(q, Qb, (int)(HEADSZ / 8));
  cvt_kernel<<<2048, blk, 0, stream>>>(k, Kb, (int)(HEADSZ / 8));
  cvt_kernel<<<2048, blk, 0, stream>>>(v, Vb, (int)(HEADSZ / 8));
  cvt_kernel<<<512,  blk, 0, stream>>>(wq_w, Wqb, (int)(WSZ / 8));
  cvt_kernel<<<512,  blk, 0, stream>>>(wk_w, Wkb, (int)(WSZ / 8));
  cvt_kernel<<<512,  blk, 0, stream>>>(wv_w, Wvb, (int)(WSZ / 8));
  cvt_kernel<<<512,  blk, 0, stream>>>(dense_w, Wdb, (int)(WSZ / 8));

  proj_kernel<<<dim3(16, 32), blk, 0, stream>>>(Qb, Wqb, wq_b, Qh, 0);
  proj_kernel<<<dim3(16, 32), blk, 0, stream>>>(Kb, Wkb, wk_b, Kh, 0);
  proj_kernel<<<dim3(16, 32), blk, 0, stream>>>(Vb, Wvb, wv_b, Vt, 1);

  size_t lds = (32 * 2048 + 256 + 32) * sizeof(float);  // 263,296 B (< 320 KB)
  attn_kernel<<<dim3(SS / 32, BB * HH), blk, lds, stream>>>(Qh, Kh, Vt, mask,
                                                            attn, Ctx);

  dense_kernel<<<dim3(16, 32), blk, 0, stream>>>(Ctx, Wdb, dense_b, out);
}
